// HSTGNN_7172595384466
// MI455X (gfx1250) — compile-verified, matrix operand order checked by reading, not run on hardware
//
#include <hip/hip_runtime.h>
#include <hip/hip_bf16.h>
#include <cstdint>
#include <cstddef>

// ---------------------------------------------------------------------------
// Types for CDNA5 WMMA
// ---------------------------------------------------------------------------
typedef __attribute__((ext_vector_type(16))) _Float16 v16h;
typedef __attribute__((ext_vector_type(8)))  _Float16 v8h;
typedef __attribute__((ext_vector_type(8)))  float    v8f;

__device__ __forceinline__ v8f wmma_f16(v16h a, v16h b, v8f c) {
    // D = A(16x32 f16) * B(32x16 f16) + C(16x16 f32)
    return __builtin_amdgcn_wmma_f32_16x16x32_f16(
        /*neg_a=*/false, a, /*neg_b=*/false, b,
        /*c_mod=*/(short)0, c, /*reuse_a=*/false, /*reuse_b=*/false);
}

// A/B fragment gather from an LDS row-major [row][PAD] f16 tile.
// lane<16 holds K {0..7,16..23}, lane>=16 holds K {8..15,24..31} (kb = 0 or 8).
__device__ __forceinline__ v16h frag_from_lds(const _Float16* p, int kb) {
    union { v16h v; v8h h[2]; } u;
    u.h[0] = *(const v8h*)(p + kb);
    u.h[1] = *(const v8h*)(p + kb + 16);
    return u.v;
}

// Fragment built from a global row (fp32/fp16, element stride STRIDE),
// with column guard only on the partial-chunk path.
template <typename T, int STRIDE>
__device__ __forceinline__ v16h frag_from_row(const T* __restrict__ row,
                                              int cbase, int kb, int limit,
                                              bool valid) {
    v16h a;
    if (valid && (cbase + 31 < limit)) {
#pragma unroll
        for (int i = 0; i < 8; ++i) {
            a[i]     = (_Float16)(float)row[(size_t)(cbase + kb + i) * STRIDE];
            a[i + 8] = (_Float16)(float)row[(size_t)(cbase + kb + 16 + i) * STRIDE];
        }
    } else {
#pragma unroll
        for (int i = 0; i < 8; ++i) {
            int c0 = cbase + kb + i;
            int c1 = cbase + kb + 16 + i;
            a[i]     = (_Float16)((valid && c0 < limit) ? (float)row[(size_t)c0 * STRIDE] : 0.f);
            a[i + 8] = (_Float16)((valid && c1 < limit) ? (float)row[(size_t)c1 * STRIDE] : 0.f);
        }
    }
    return a;
}

// sigmoid via v_exp_f32 + v_rcp_f32 only (no IEEE div chain)
__device__ __forceinline__ float fast_sigmoid(float u) {
    return __builtin_amdgcn_rcpf(1.f + __expf(-u));
}
__device__ __forceinline__ float elu_f(float v) {
    return v > 0.f ? v : (__expf(v) - 1.f);
}

// ---------------------------------------------------------------------------
// Constants
// ---------------------------------------------------------------------------
#define BATCH 8
#define CH    32
#define NNODE 1000
#define TSEQ  336
#define LD    336   // time stride of fp32 (B,C,N,T) buffers AND f16 (B,N,T,C)
#define PAD   40    // LDS tile row pitch in halves (80B: 16B-aligned, no bank conflicts)

struct IncPtrs {
    const float* fw[4];
    const float* fb[4];
    const float* gw[4];
    const float* gb[4];
};

// ---------------------------------------------------------------------------
// Kernel 1: dilated inception (4 branches x (filter|gate)) + tanh*sigmoid gate
// One block per (n, b). LDS: full time series tile + all tap weights.
// Output xg is f16 in (B, N, T, C) layout (contiguous (t,c) slice per (b,n)).
// ---------------------------------------------------------------------------
__global__ void __launch_bounds__(256)
inception_gate_kernel(const float* __restrict__ src, int t_in, int t_out,
                      int dil, IncPtrs P, _Float16* __restrict__ xg) {
    const int KER[4] = {2, 3, 6, 7};
    const int TB[4]  = {0, 2, 5, 11};   // tap bases, total 18 taps

    __shared__ __align__(16) _Float16 sx[352 * PAD];      // [t][ci]
    __shared__ __align__(16) _Float16 sw[18 * 16 * PAD];  // [tap][co'][ci]
    __shared__ float sb[64];                              // [branch*16+co']

    const int tid = threadIdx.x;
    const int n = blockIdx.x, b = blockIdx.y;

    // Stage input time series: consecutive lanes read consecutive t (coalesced).
    for (int ci = 0; ci < CH; ++ci) {
        const float* srow = src + (((size_t)b * CH + ci) * NNODE + n) * LD;
        for (int t = tid; t < 352; t += 256)
            sx[t * PAD + ci] = (t < t_in) ? (_Float16)srow[t] : (_Float16)0.f;
    }
    // Stage combined filter|gate weights, transposed to [tap][co'][ci].
    for (int idx = tid; idx < 18 * 512; idx += 256) {
        int tap = idx >> 9;
        int rem = idx & 511;
        int co = rem >> 5, ci = rem & 31;
        int br = (tap >= 11) ? 3 : (tap >= 5) ? 2 : (tap >= 2) ? 1 : 0;
        int kk = KER[br], kap = tap - TB[br];
        float w = (co < 8) ? P.fw[br][(co * CH + ci) * kk + kap]
                           : P.gw[br][((co - 8) * CH + ci) * kk + kap];
        sw[(tap * 16 + co) * PAD + ci] = (_Float16)w;
    }
    if (tid < 64) {
        int br = tid >> 4, co = tid & 15;
        sb[tid] = (co < 8) ? P.fb[br][co] : P.gb[br][co - 8];
    }
    __syncthreads();

    const int wave = tid >> 5, lane = tid & 31;
    const int mrow = lane & 15;
    const int kb = (lane < 16) ? 0 : 8;
    const int ntile = (t_out + 15) >> 4;

    for (int item = wave; item < ntile * 4; item += 8) {
        int br = item & 3, tile = item >> 2;
        int k = KER[br];
        int t0 = tile << 4;
        int abase = t0 + mrow + (7 - k) * dil;   // branch alignment offset

        v8f c = {};
        for (int kap = 0; kap < k; ++kap) {
            v16h a = frag_from_lds(&sx[(abase + kap * dil) * PAD], kb);
            v16h w = frag_from_lds(&sw[((TB[br] + kap) * 16 + mrow) * PAD], kb);
            c = wmma_f16(a, w, c);
        }

        int ncol = lane & 15;
        bool isF = ncol < 8;                 // cols 0-7 filter, 8-15 gate
        float bias = sb[br * 16 + ncol];
        // (B,N,T,C) output: element (b,n,j,ch)
        size_t obase = ((size_t)(b * NNODE + n)) * LD * CH + (br * 8 + (ncol & 7));
        float fscl = isF ? 2.f : 1.f;        // tanh(v) = 2*sigmoid(2v) - 1
        float fsub = isF ? 1.f : 0.f;
#pragma unroll
        for (int r = 0; r < 8; ++r) {
            int j = t0 + r + ((lane < 16) ? 0 : 8);
            float v = c[r] + bias;
            float act = fscl * fast_sigmoid(fscl * v) - fsub;  // branchless
            float other = __shfl_xor(act, 8, 32);
            if (isF && j < t_out)
                xg[obase + (size_t)j * CH] = (_Float16)(act * other);
        }
    }
}

// ---------------------------------------------------------------------------
// Kernel 2: skip conv (full C x T reduction -> T=1), accumulate into skip.
// Block = (16 nodes, one b); M=nodes, N=32 co (2 tiles), K=32ci x t-chunks.
// 8 waves split the ci dimension; LDS reduction across waves.
// NTC=false: src fp32 (B,C,N,T);  NTC=true: src f16 (B,N,T,C).
// ---------------------------------------------------------------------------
template <typename T, bool NTC>
__global__ void __launch_bounds__(256)
skip_conv_kernel(const T* __restrict__ src, int tlen,
                 const float* __restrict__ w, const float* __restrict__ bias,
                 float* __restrict__ skip, int init) {
    __shared__ float red[2 * 8 * 32 * 8];   // [cotile][wave][lane][r]

    const int tid = threadIdx.x, wave = tid >> 5, lane = tid & 31;
    const int n0 = blockIdx.x * 16, b = blockIdx.y;
    const int m = lane & 15, kb = (lane < 16) ? 0 : 8;
    const int node = n0 + m;
    const bool nvalid = node < NNODE;
    const int nd_safe = nvalid ? node : 0;
    const int nchunk = (tlen + 31) >> 5;
    constexpr int AS = NTC ? CH : 1;        // A column (time) stride

    v8f c0 = {}, c1 = {};
    for (int ci = wave * 4; ci < wave * 4 + 4; ++ci) {
        const T* arow = NTC
            ? src + ((size_t)(b * NNODE + nd_safe)) * LD * CH + ci
            : src + (((size_t)b * CH + ci) * NNODE + nd_safe) * LD;
        const float* b0 = w + ((size_t)m * CH + ci) * tlen;
        const float* b1 = w + ((size_t)(m + 16) * CH + ci) * tlen;
        for (int tc = 0; tc < nchunk; ++tc) {
            int cb = tc << 5;
            __builtin_prefetch(b0 + cb + 32, 0, 0);
            v16h a  = frag_from_row<T, AS>(arow, cb, kb, tlen, nvalid);
            v16h w0 = frag_from_row<float, 1>(b0, cb, kb, tlen, true);
            v16h w1 = frag_from_row<float, 1>(b1, cb, kb, tlen, true);
            c0 = wmma_f16(a, w0, c0);
            c1 = wmma_f16(a, w1, c1);
        }
    }
#pragma unroll
    for (int r = 0; r < 8; ++r) {
        red[((0 * 8 + wave) * 32 + lane) * 8 + r] = c0[r];
        red[((1 * 8 + wave) * 32 + lane) * 8 + r] = c1[r];
    }
    __syncthreads();

    for (int f = tid; f < 512; f += 256) {
        int ct = f >> 8;
        int ln = (f >> 3) & 31;
        int r = f & 7;
        float s = 0.f;
#pragma unroll
        for (int w8 = 0; w8 < 8; ++w8)
            s += red[((ct * 8 + w8) * 32 + ln) * 8 + r];
        int mm = r + ((ln < 16) ? 0 : 8);
        int co = (ln & 15) + 16 * ct;
        int nd = n0 + mm;
        if (nd < NNODE) {
            size_t oi = ((size_t)b * CH + co) * NNODE + nd;
            float prev = init ? 0.f : skip[oi];
            skip[oi] = prev + s + bias[co];
        }
    }
}

// ---------------------------------------------------------------------------
// Kernel 3: 1x1 res conv + residual add + LayerNorm partial sums (per b).
// One block per (n, b); M=16 t, K=32 ci, N=32 co (2 tiles).
// xg (B,N,T,C) slice is contiguous -> staged with async global->LDS copies.
// ---------------------------------------------------------------------------
__global__ void __launch_bounds__(256)
res_norm_kernel(const _Float16* __restrict__ xg, int t_out,
                const float* __restrict__ rw, const float* __restrict__ rb,
                const float* __restrict__ xprev, int t_prev,
                float* __restrict__ xnext, float* __restrict__ stats) {
    __shared__ __align__(16) _Float16 sx[336 * PAD];   // [t][ci]
    __shared__ float rs1[256], rs2[256];

    const int tid = threadIdx.x;
    const int n = blockIdx.x, b = blockIdx.y;
    const int ntile = (t_out + 15) >> 4, tpad = ntile << 4;

    // ---- async stage: (t_out x 32ch) f16 slice, 16B chunks, ASYNCcnt-tracked
    {
        const _Float16* g = xg + (size_t)(b * NNODE + n) * LD * CH;
        const int nchunk = t_out * 4;              // 4 x 16B per 64B row
        for (int idx = tid; idx < nchunk; idx += 256) {
            int t = idx >> 2, c4 = idx & 3;
            uint32_t lds_addr = (uint32_t)(uintptr_t)&sx[t * PAD + c4 * 8];
            const _Float16* ga = g + (size_t)t * CH + c4 * 8;
            asm volatile("global_load_async_to_lds_b128 %0, %1, off"
                         :: "v"(lds_addr), "v"(ga) : "memory");
        }
        // zero-pad rows [t_out, tpad)
        v8h z = {};
        for (int idx = t_out * 4 + tid; idx < tpad * 4; idx += 256) {
            int t = idx >> 2, c4 = idx & 3;
            *(v8h*)&sx[t * PAD + c4 * 8] = z;
        }
        asm volatile("s_wait_asynccnt 0x0" ::: "memory");
    }
    __syncthreads();

    const int wave = tid >> 5, lane = tid & 31;
    const int m = lane & 15, kb = (lane < 16) ? 0 : 8;
    const int off = t_prev - t_out;   // residual truncation offset (= 6*d)

    // B fragments (rw is 32x32, row-major (co,ci)) — invariant across tiles.
    v16h w0 = frag_from_row<float, 1>(rw + m * CH, 0, kb, CH, true);
    v16h w1 = frag_from_row<float, 1>(rw + (m + 16) * CH, 0, kb, CH, true);

    float s1 = 0.f, s2 = 0.f;
    for (int tile = wave; tile < ntile; tile += 8) {
        int t0 = tile << 4;
        v16h a = frag_from_lds(&sx[(t0 + m) * PAD], kb);
        v8f c0 = {}, c1 = {};
        c0 = wmma_f16(a, w0, c0);
        c1 = wmma_f16(a, w1, c1);
#pragma unroll
        for (int ct = 0; ct < 2; ++ct) {
            v8f c = ct ? c1 : c0;
            int co = m + 16 * ct;
            size_t rbase = ((size_t)b * CH + co) * NNODE + n;
#pragma unroll
            for (int r = 0; r < 8; ++r) {
                int j = t0 + r + ((lane < 16) ? 0 : 8);
                if (j < t_out) {
                    float y = c[r] + rb[co] + xprev[rbase * LD + off + j];
                    xnext[rbase * LD + j] = y;
                    s1 += y;
                    s2 += y * y;
                }
            }
        }
    }

    rs1[tid] = s1;
    rs2[tid] = s2;
    __syncthreads();
    for (int s = 128; s > 0; s >>= 1) {
        if (tid < s) { rs1[tid] += rs1[tid + s]; rs2[tid] += rs2[tid + s]; }
        __syncthreads();
    }
    if (tid == 0) {
        atomicAdd(&stats[b * 2 + 0], rs1[0]);
        atomicAdd(&stats[b * 2 + 1], rs2[0]);
    }
}

// ---------------------------------------------------------------------------
// Kernel 4: LayerNorm apply (elementwise, full affine over (C,N,T))
// ---------------------------------------------------------------------------
__global__ void __launch_bounds__(256)
ln_apply_kernel(float* __restrict__ xv, int t_out,
                const float* __restrict__ nw, const float* __restrict__ nb,
                const float* __restrict__ stats) {
    size_t total = (size_t)BATCH * CH * NNODE * t_out;
    float cnt = (float)CH * NNODE * t_out;
    for (size_t idx = (size_t)blockIdx.x * 256 + threadIdx.x; idx < total;
         idx += (size_t)gridDim.x * 256) {
        int j = (int)(idx % t_out);
        size_t r = idx / t_out;
        int nn = (int)(r % NNODE); r /= NNODE;
        int cc = (int)(r % CH);
        int bb = (int)(r / CH);
        float mu = stats[bb * 2] / cnt;
        float var = stats[bb * 2 + 1] / cnt - mu * mu;
        float inv = rsqrtf(var + 1e-5f);
        size_t xi = (((size_t)bb * CH + cc) * NNODE + nn) * LD + j;
        size_t pi = ((size_t)cc * NNODE + nn) * (size_t)t_out + j;
        float y = (xv[xi] - mu) * inv;
        xv[xi] = y * nw[pi] + nb[pi];
    }
}

// ---------------------------------------------------------------------------
// Kernel 5: out = elu( end_w * elu(skip) + end_b )
// ---------------------------------------------------------------------------
__global__ void __launch_bounds__(256)
end_kernel(const float* __restrict__ skip, const float* __restrict__ ew,
           const float* __restrict__ eb, float* __restrict__ out) {
    int tid = blockIdx.x * 256 + threadIdx.x;
    if (tid >= BATCH * NNODE * CH) return;
    int co = tid & 31;
    int bn = tid >> 5;
    int b = bn / NNODE, n = bn - b * NNODE;
    float s = eb[co];
#pragma unroll 4
    for (int ci = 0; ci < CH; ++ci) {
        float v = elu_f(skip[((size_t)b * CH + ci) * NNODE + n]);
        s += ew[co * CH + ci] * v;
    }
    out[((size_t)b * CH + co) * NNODE + n] = elu_f(s);
}

__global__ void zero_stats_kernel(float* __restrict__ stats) {
    if (threadIdx.x < 16) stats[threadIdx.x] = 0.f;
}

// ---------------------------------------------------------------------------
// Host launch
// ---------------------------------------------------------------------------
extern "C" void kernel_launch(void* const* d_in, const int* in_sizes, int n_in,
                              void* d_out, int out_size, void* d_ws,
                              size_t ws_size, hipStream_t stream) {
    (void)in_sizes; (void)n_in; (void)out_size; (void)ws_size;

    const int DIL[5]  = {1, 2, 4, 8, 16};
    const int TOUT[5] = {330, 318, 294, 246, 150};

    const float* x       = (const float*)d_in[0];
    const float* skip0_w = (const float*)d_in[1];
    const float* skip0_b = (const float*)d_in[2];
    const float* skipE_w = (const float*)d_in[3];
    const float* skipE_b = (const float*)d_in[4];
    const float* end_w   = (const float*)d_in[5];
    const float* end_b   = (const float*)d_in[6];

    // Workspace carve-up
    uint8_t* ws = (uint8_t*)d_ws;
    size_t off = 0;
    auto alloc = [&](size_t bytes) {
        void* p = ws + off;
        off += (bytes + 255) & ~(size_t)255;
        return p;
    };
    const size_t XB = (size_t)BATCH * CH * NNODE * LD;
    float*     xa    = (float*)alloc(XB * sizeof(float));
    float*     xb    = (float*)alloc(XB * sizeof(float));
    _Float16*  xg    = (_Float16*)alloc(XB * sizeof(_Float16));
    float*     skip  = (float*)alloc((size_t)BATCH * CH * NNODE * sizeof(float));
    float*     stats = (float*)alloc(64);

    const dim3 blk(256);
    const dim3 grid_bn(NNODE, BATCH);             // one block per (n, b)
    const dim3 grid_nt((NNODE + 15) / 16, BATCH); // 16-node tiles per b

    // skip = conv(x, skip0)  (T=336 -> 1, initializes skip accumulator)
    skip_conv_kernel<float, false><<<grid_nt, blk, 0, stream>>>(
        x, TSEQ, skip0_w, skip0_b, skip, 1);

    const float* xcur = x;
    int t_in = TSEQ;
    float* pingpong[2] = {xa, xb};
    int pb = 0;

    for (int j = 0; j < 5; ++j) {
        const int base = 7 + 22 * j;
        IncPtrs P;
        for (int i = 0; i < 4; ++i) {
            P.fw[i] = (const float*)d_in[base + i];
            P.fb[i] = (const float*)d_in[base + 4 + i];
            P.gw[i] = (const float*)d_in[base + 8 + i];
            P.gb[i] = (const float*)d_in[base + 12 + i];
        }
        const float* res_w  = (const float*)d_in[base + 16];
        const float* res_b  = (const float*)d_in[base + 17];
        const float* skp_w  = (const float*)d_in[base + 18];
        const float* skp_b  = (const float*)d_in[base + 19];
        const float* norm_w = (const float*)d_in[base + 20];
        const float* norm_b = (const float*)d_in[base + 21];

        const int d = DIL[j], t_out = TOUT[j];
        float* xnext = pingpong[pb];

        inception_gate_kernel<<<grid_bn, blk, 0, stream>>>(
            xcur, t_in, t_out, d, P, xg);

        skip_conv_kernel<_Float16, true><<<grid_nt, blk, 0, stream>>>(
            xg, t_out, skp_w, skp_b, skip, 0);

        zero_stats_kernel<<<1, 32, 0, stream>>>(stats);

        res_norm_kernel<<<grid_bn, blk, 0, stream>>>(
            xg, t_out, res_w, res_b, xcur, t_in, xnext, stats);

        size_t total = (size_t)BATCH * CH * NNODE * t_out;
        int ln_blocks = (int)((total + 255) / 256);
        ln_apply_kernel<<<ln_blocks, blk, 0, stream>>>(
            xnext, t_out, norm_w, norm_b, stats);

        xcur = xnext;
        pb ^= 1;
        t_in = t_out;
    }

    // skip += conv(x, skipE)  (T=150 -> 1)
    skip_conv_kernel<float, false><<<grid_nt, blk, 0, stream>>>(
        xcur, TOUT[4], skipE_w, skipE_b, skip, 0);

    // out = elu(end_w * elu(skip) + end_b)
    int nthreads = BATCH * NNODE * CH;
    end_kernel<<<(nthreads + 255) / 256, blk, 0, stream>>>(
        skip, end_w, end_b, (float*)d_out);
}
